// TreeEncoder_67611375173685
// MI455X (gfx1250) — matrix operand order repaired and checked by reading, not compiled, not run on hardware
//
#include <hip/hip_runtime.h>

// ---------------------------------------------------------------------------
// TreeEncoder fused kernel for gfx1250 (MI455X): embedding GEMM + 2 tree-LSTM
// layers + mean over nodes, one pass over tree_features (HBM floor ~5.8us).
// All 7 GEMMs run on v_wmma_f32_16x16x32_f16; activations use fast
// rcp/exp2 hardware transcendentals; biases are folded into the WMMA C init.
// ---------------------------------------------------------------------------

typedef __attribute__((ext_vector_type(16))) _Float16 v16h;
typedef __attribute__((ext_vector_type(8)))  _Float16 v8h;
typedef __attribute__((ext_vector_type(8)))  float    v8f;
typedef __attribute__((ext_vector_type(4)))  float    v4f;

#define B_  64
#define N_  8192
#define F_  64
#define H_  64
#define L_  2

// LDS layout (halves). Row stride 72 (=9*8) keeps 16B alignment for b128 ops
// while breaking the 128B power-of-two bank pattern.
constexpr int WROW  = 72;
constexpr int WMAT  = 64 * WROW;          // one 64x64 f16 matrix, padded
constexpr int NMAT  = 7;                  // W_emb + 2 layers * 3 gates
constexpr int HTILE = 16 * WROW;          // per-wave 16x64 h tile, padded
constexpr int WAVES = 8;                  // 256 threads
constexpr int NODES_PER_ITER  = WAVES * 16;              // 128
constexpr int ITERS           = 16;
constexpr int NODES_PER_BLOCK = NODES_PER_ITER * ITERS;  // 2048
constexpr int BLOCKS_PER_BATCH = N_ / NODES_PER_BLOCK;   // 4
constexpr int LDS_HALVES = NMAT * WMAT + WAVES * HTILE;  // -> 82944 B dynamic

__device__ __forceinline__ v8f splat8(float v) {
  v8f z;
#pragma unroll
  for (int i = 0; i < 8; ++i) z[i] = v;
  return z;
}

__device__ __forceinline__ v8f wmma16(v16h a, v16h b, v8f c) {
  // D = A(16x32 f16) * B(32x16 f16) + C(16x16 f32)
  return __builtin_amdgcn_wmma_f32_16x16x32_f16(
      /*neg_a=*/false, a, /*neg_b=*/false, b,
      /*c_mod=*/(short)0, c, /*reuse_a=*/false, /*reuse_b=*/false);
}

// A-fragment (16x32 f16) straight from global fp32 rows.
// lanes 0-15: row = lane,    halves = K {k0..k0+7, k0+16..k0+23}, k0 = 32*ck
// lanes 16-31: row = lane&15, k0 = 32*ck + 8
__device__ __forceinline__ v16h make_a_global(const float* __restrict__ rowp, int k0) {
  v4f f0 = *(const v4f*)(rowp + k0);
  v4f f1 = *(const v4f*)(rowp + k0 + 4);
  v4f f2 = *(const v4f*)(rowp + k0 + 16);
  v4f f3 = *(const v4f*)(rowp + k0 + 20);
  v16h a;
#pragma unroll
  for (int i = 0; i < 4; ++i) {
    a[i]      = (_Float16)f0[i];
    a[4 + i]  = (_Float16)f1[i];
    a[8 + i]  = (_Float16)f2[i];
    a[12 + i] = (_Float16)f3[i];
  }
  return a;
}

// A-fragment from the wave's LDS h tile (f16, row-major, stride WROW).
__device__ __forceinline__ v16h make_a_lds(const _Float16* hb, int ck, int lrow, int khalf8) {
  const _Float16* p = hb + lrow * WROW + ck * 32 + khalf8;   // 16B aligned
  v8h lo = *(const v8h*)p;
  v8h hi = *(const v8h*)(p + 16);
  return __builtin_shufflevector(lo, hi, 0, 1, 2, 3, 4, 5, 6, 7,
                                          8, 9, 10, 11, 12, 13, 14, 15);
}

// B-fragment (32x16 f16): lane holds column o = t*16 + (lane&15),
// K halves {k0..k0+7, k0+16..k0+23}, k0 = ck*32 + (lane>=16 ? 8 : 0).
__device__ __forceinline__ v16h make_b_lds(const _Float16* sm, int mat, int t,
                                           int ck, int lrow, int khalf16) {
  const _Float16* p = sm + mat * WMAT + (t * 16 + lrow) * WROW + ck * 32 + (khalf16 >> 1);
  v8h lo = *(const v8h*)p;
  v8h hi = *(const v8h*)(p + 16);
  return __builtin_shufflevector(lo, hi, 0, 1, 2, 3, 4, 5, 6, 7,
                                          8, 9, 10, 11, 12, 13, 14, 15);
}

// Fast activations on the hardware transcendental pipe:
// sigmoid(x) = rcp(1 + 2^(-x*log2e)); tanh(x) = 1 - 2*rcp(1 + 2^(2x*log2e))
__device__ __forceinline__ float fast_sigmoid(float x) {
  return __builtin_amdgcn_rcpf(1.0f + __builtin_amdgcn_exp2f(-1.44269504f * x));
}
__device__ __forceinline__ float fast_tanh(float x) {
  return 1.0f - 2.0f * __builtin_amdgcn_rcpf(1.0f + __builtin_amdgcn_exp2f(2.88539008f * x));
}

__global__ void TreeEncoder_zero(float* __restrict__ out) {
  int i = blockIdx.x * blockDim.x + threadIdx.x;
  if (i < B_ * H_) out[i] = 0.0f;
}

__launch_bounds__(256)
__global__ void TreeEncoder_fused(const float* __restrict__ x,     // [B,N,F]
                                  const float* __restrict__ Wemb,  // [H,F]
                                  const float* __restrict__ bemb,  // [H]
                                  const float* __restrict__ Wx,    // [L,3,H,H]
                                  const float* __restrict__ bW,    // [L,3,H]
                                  const float* __restrict__ bU,    // [L,3,H]
                                  float* __restrict__ out) {       // [B,H]
  extern __shared__ _Float16 smem[];

  const int tid  = threadIdx.x;
  const int wave = tid >> 5;
  const int lane = tid & 31;
  const int b    = blockIdx.x;
  const int tg   = blockIdx.y;

  // ---- stage all 7 weight matrices to LDS as f16 (padded rows) ----
  for (int idx = tid; idx < NMAT * 4096; idx += 256) {
    int m = idx >> 12;
    int e = idx & 4095;
    int row = e >> 6, col = e & 63;
    float w = (m == 0) ? Wemb[e] : Wx[(m - 1) * 4096 + e];
    smem[m * WMAT + row * WROW + col] = (_Float16)w;
  }
  __syncthreads();

  const int  lrow    = lane & 15;
  const bool hiHalf  = lane >= 16;
  const int  khalf8  = hiHalf ? 8 : 0;    // A-fragment K base within chunk
  const int  khalf16 = hiHalf ? 16 : 0;   // B-fragment K base within chunk
  const int  rbase   = hiHalf ? 8 : 0;    // D-fragment row base

  // ---- per-lane biases for each 16-wide H coltile (column == lane dim) ----
  float bz_emb[4];
  float bz[L_][3][4];
#pragma unroll
  for (int t = 0; t < 4; ++t) {
    int h = t * 16 + lrow;
    bz_emb[t] = bemb[h];
#pragma unroll
    for (int l = 0; l < L_; ++l)
#pragma unroll
      for (int g = 0; g < 3; ++g)
        bz[l][g][t] = bW[(l * 3 + g) * 64 + h] + bU[(l * 3 + g) * 64 + h];
  }

  // ---- hoist the loop-invariant embedding B-fragments into registers ----
  v16h embB[2][4];
#pragma unroll
  for (int ck = 0; ck < 2; ++ck)
#pragma unroll
    for (int t = 0; t < 4; ++t)
      embB[ck][t] = make_b_lds(smem, 0, t, ck, lrow, khalf16);

  _Float16* hb = &smem[NMAT * WMAT + wave * HTILE];
  float acc[4] = {0.f, 0.f, 0.f, 0.f};

  for (int it = 0; it < ITERS; ++it) {
    const int node0 = tg * NODES_PER_BLOCK + it * NODES_PER_ITER + wave * 16;
    const float* rowp = x + ((size_t)b * N_ + node0 + lrow) * (size_t)F_;
    if (it + 1 < ITERS)
      __builtin_prefetch(rowp + (size_t)NODES_PER_ITER * F_, 0, 0);

    // -------- embedding: h0 = X * Wemb^T + bemb (bias as C init) --------
    v16h ag0 = make_a_global(rowp, khalf8);
    v16h ag1 = make_a_global(rowp, 32 + khalf8);
    v8f h0[4];
#pragma unroll
    for (int t = 0; t < 4; ++t) {
      v8f z = splat8(bz_emb[t]);
      z = wmma16(ag0, embB[0][t], z);
      h0[t] = wmma16(ag1, embB[1][t], z);
    }
    // store h0 (f16) to this wave's LDS tile (D-layout -> row-major tile)
#pragma unroll
    for (int t = 0; t < 4; ++t) {
      int col = t * 16 + lrow;
#pragma unroll
      for (int r = 0; r < 8; ++r)
        hb[(rbase + r) * WROW + col] = (_Float16)h0[t][r];
    }
    asm volatile("s_wait_dscnt 0" ::: "memory");

    // ---------------- two tree-LSTM layers ----------------
#pragma unroll
    for (int l = 0; l < L_; ++l) {
      v16h a0 = make_a_lds(hb, 0, lrow, khalf8);
      v16h a1 = make_a_lds(hb, 1, lrow, khalf8);

      v8f ig[4], og[4];
      // ---- input gate: bulk-load all 8 B fragments, then 8 WMMAs ----
      {
        v16h b0[4], b1[4];
#pragma unroll
        for (int t = 0; t < 4; ++t) {
          b0[t] = make_b_lds(smem, 1 + l * 3 + 0, t, 0, lrow, khalf16);
          b1[t] = make_b_lds(smem, 1 + l * 3 + 0, t, 1, lrow, khalf16);
        }
#pragma unroll
        for (int t = 0; t < 4; ++t) {
          v8f z = splat8(bz[l][0][t]);
          z = wmma16(a0, b0[t], z);
          z = wmma16(a1, b1[t], z);
#pragma unroll
          for (int r = 0; r < 8; ++r) ig[t][r] = fast_sigmoid(z[r]);
        }
      }
      // ---- output gate ----
      {
        v16h b0[4], b1[4];
#pragma unroll
        for (int t = 0; t < 4; ++t) {
          b0[t] = make_b_lds(smem, 1 + l * 3 + 1, t, 0, lrow, khalf16);
          b1[t] = make_b_lds(smem, 1 + l * 3 + 1, t, 1, lrow, khalf16);
        }
#pragma unroll
        for (int t = 0; t < 4; ++t) {
          v8f z = splat8(bz[l][1][t]);
          z = wmma16(a0, b0[t], z);
          z = wmma16(a1, b1[t], z);
#pragma unroll
          for (int r = 0; r < 8; ++r) og[t][r] = fast_sigmoid(z[r]);
        }
      }
      // ---- cell candidate + combine: h = o * tanh(i * tanh(zc)) ----
      {
        v16h b0[4], b1[4];
#pragma unroll
        for (int t = 0; t < 4; ++t) {
          b0[t] = make_b_lds(smem, 1 + l * 3 + 2, t, 0, lrow, khalf16);
          b1[t] = make_b_lds(smem, 1 + l * 3 + 2, t, 1, lrow, khalf16);
        }
#pragma unroll
        for (int t = 0; t < 4; ++t) {
          v8f z = splat8(bz[l][2][t]);
          z = wmma16(a0, b0[t], z);
          z = wmma16(a1, b1[t], z);
          int col = t * 16 + lrow;
#pragma unroll
          for (int r = 0; r < 8; ++r) {
            float ct = fast_tanh(z[r]);
            float c  = ig[t][r] * ct;
            float h  = og[t][r] * fast_tanh(c);
            if (l == 0)
              hb[(rbase + r) * WROW + col] = (_Float16)h;  // feed layer 1
            else
              acc[t] += h;                                  // mean accumulator
          }
        }
      }
      if (l == 0) asm volatile("s_wait_dscnt 0" ::: "memory");
    }
  }

  // ---- reduction: rows 0-7 live in lanes 0-15, rows 8-15 in lanes 16-31;
  //      both lane halves atomically add into out[b, t*16 + lrow]. ----
  const float scale = 1.0f / (float)N_;
#pragma unroll
  for (int t = 0; t < 4; ++t)
    atomicAdd(&out[b * H_ + t * 16 + lrow], acc[t] * scale);
}

extern "C" void kernel_launch(void* const* d_in, const int* in_sizes, int n_in,
                              void* d_out, int out_size, void* d_ws, size_t ws_size,
                              hipStream_t stream) {
  (void)in_sizes; (void)n_in; (void)d_ws; (void)ws_size; (void)out_size;
  const float* x    = (const float*)d_in[0];
  const float* Wemb = (const float*)d_in[1];
  const float* bemb = (const float*)d_in[2];
  const float* Wx   = (const float*)d_in[3];
  const float* bW   = (const float*)d_in[4];
  const float* bU   = (const float*)d_in[5];
  float* out = (float*)d_out;

  TreeEncoder_zero<<<dim3((B_ * H_ + 255) / 256), dim3(256), 0, stream>>>(out);

  dim3 grid(B_, BLOCKS_PER_BATCH);
  size_t lds_bytes = (size_t)LDS_HALVES * sizeof(_Float16);  // 82944 B
  TreeEncoder_fused<<<grid, dim3(256), lds_bytes, stream>>>(
      x, Wemb, bemb, Wx, bW, bU, out);
}
